// HardCausalTemporalEncoder_21689584845617
// MI455X (gfx1250) — matrix operand order repaired and checked by reading, not compiled, not run on hardware
//
#include <hip/hip_runtime.h>
#include <hip/hip_bf16.h>
#include <math.h>
#include <stdint.h>

typedef __attribute__((ext_vector_type(16))) __bf16 v16bf;
typedef __attribute__((ext_vector_type(8)))  float  v8f;

#define N_TOK 4096
#define EDIM  512
#define NHEAD 8
#define HD    64

__device__ __forceinline__ v8f wmma_bf16(v16bf a, v16bf b, v8f c) {
    return __builtin_amdgcn_wmma_f32_16x16x32_bf16(false, a, false, b, (short)0, c, false, false);
}

// ---- CDNA5 async global->LDS (ASYNCcnt) ------------------------------------
__device__ __forceinline__ uint32_t lds_off(const void* p) {
    return (uint32_t)(uintptr_t)p;            // flat->LDS keeps offset in addr[31:0]
}
__device__ __forceinline__ void async_copy_b128(uint32_t lds_addr, const void* g) {
    asm volatile("global_load_async_to_lds_b128 %0, %1, off"
                 :: "v"(lds_addr), "v"(g) : "memory");
}
__device__ __forceinline__ void wait_async0() {
    asm volatile("s_wait_asynccnt 0x0" ::: "memory");
}

// ---- fragment loaders -------------------------------------------------------
// A fragment (16x32, MxK): lane = g*16+nl holds row m=nl,
//   elems 0..15 -> K = k0 + g*16 + e (contiguous)
__device__ __forceinline__ v16bf load_a_f32(const float* __restrict__ A, int lda,
                                            int row0, int k0, int lane) {
    int nl = lane & 15, g = lane >> 4;
    const float* r = A + (size_t)(row0 + nl) * lda + k0;
    v16bf a;
#pragma unroll
    for (int e = 0; e < 8; ++e) { a[e]     = (__bf16)r[g * 16 + e]; }
#pragma unroll
    for (int e = 0; e < 8; ++e) { a[8 + e] = (__bf16)r[g * 16 + 8 + e]; }
    return a;
}

__device__ __forceinline__ v16bf load_a_bf(const __bf16* __restrict__ A, int lda,
                                           int row0, int k0, int lane) {
    int nl = lane & 15, g = lane >> 4;
    const __bf16* r = A + (size_t)(row0 + nl) * lda + k0;
    v16bf a;
#pragma unroll
    for (int e = 0; e < 8; ++e) { a[e]     = r[g * 16 + e]; }
#pragma unroll
    for (int e = 0; e < 8; ++e) { a[8 + e] = r[g * 16 + 8 + e]; }
    return a;
}

// B fragment (32x16, KxN) with B = W^T, W row-major [Nout, K]:
// lane = g*16+nl holds col n=nl, K = k0 + g*16 + 0..15 (contiguous).
__device__ __forceinline__ v16bf load_b_w_f32(const float* __restrict__ W, int ldw,
                                              int col0, int k0, int lane) {
    int nl = lane & 15, g = lane >> 4;
    const float* r = W + (size_t)(col0 + nl) * ldw + k0 + g * 16;
    v16bf b;
#pragma unroll
    for (int e = 0; e < 16; ++e) b[e] = (__bf16)r[e];
    return b;
}

__device__ __forceinline__ v16bf load_b_contig_bf(const __bf16* __restrict__ p) {
    v16bf b;
#pragma unroll
    for (int e = 0; e < 16; ++e) b[e] = p[e];
    return b;
}

// ============================================================================
// Kernel 1: QKV = x @ in_proj_w^T + b ; Q pre-scaled by 1/8; V stored transposed
//   4 waves/block, 4 col-tiles per wave (A fragment reused 4x)
// ============================================================================
__global__ void qkv_gemm_kernel(const float* __restrict__ x,
                                const float* __restrict__ w,
                                const float* __restrict__ bias,
                                __bf16* __restrict__ Qbf,   // [H][N][64]
                                __bf16* __restrict__ Kbf,   // [H][N][64]
                                __bf16* __restrict__ Vtbf)  // [H][64][N]
{
    int lane = threadIdx.x & 31;
    int wave = threadIdx.x >> 5;
    int row0 = blockIdx.x * 16;
    int nl = lane & 15, g = lane >> 4;

    v8f acc[4] = {v8f{}, v8f{}, v8f{}, v8f{}};
#pragma unroll 2
    for (int k0 = 0; k0 < EDIM; k0 += 32) {
        v16bf a = load_a_f32(x, EDIM, row0, k0, lane);
#pragma unroll
        for (int t = 0; t < 4; ++t) {
            int col0 = (blockIdx.y * 16 + wave * 4 + t) * 16;   // 0..1535
            v16bf b = load_b_w_f32(w, EDIM, col0, k0, lane);
            acc[t] = wmma_bf16(a, b, acc[t]);
        }
    }
#pragma unroll
    for (int t = 0; t < 4; ++t) {
        int c = (blockIdx.y * 16 + wave * 4 + t) * 16 + nl;
        float bv = bias[c];
#pragma unroll
        for (int r = 0; r < 8; ++r) {
            int row = row0 + g * 8 + r;
            float v = acc[t][r] + bv;
            if (c < 512) {                       // Q (tile stays in one section)
                int h = c >> 6, d = c & 63;
                Qbf[((size_t)h * N_TOK + row) * HD + d] = (__bf16)(v * 0.125f);
            } else if (c < 1024) {               // K
                int cc = c - 512, h = cc >> 6, d = cc & 63;
                Kbf[((size_t)h * N_TOK + row) * HD + d] = (__bf16)v;
            } else {                             // V transposed
                int cc = c - 1024, h = cc >> 6, d = cc & 63;
                Vtbf[((size_t)h * HD + d) * N_TOK + row] = (__bf16)v;
            }
        }
    }
}

// ============================================================================
// Kernel 2: flash attention. Workgroup = 8 waves on the SAME head, 128 rows.
//   K/V 32-key tiles staged in LDS via async global->LDS, double buffered.
// ============================================================================
__global__ void attention_kernel(const __bf16* __restrict__ Qbf,
                                 const __bf16* __restrict__ Kbf,
                                 const __bf16* __restrict__ Vtbf,
                                 const int* __restrict__ steps,
                                 __bf16* __restrict__ ctxbf)   // [N][512]
{
    __shared__ __bf16 kbuf[2][32 * HD];   // keys x dims, row-major
    __shared__ __bf16 vbuf[2][HD * 32];   // dims x keys (from Vt), row-major
    __shared__ __bf16 psh[8 * 16 * 32];   // per-wave P staging

    int tid  = threadIdx.x;
    int lane = tid & 31;
    int wave = tid >> 5;
    int head = blockIdx.x >> 5;                 // 8 heads x 32 row-groups
    int row0 = (blockIdx.x & 31) * 128 + wave * 16;
    int nl = lane & 15, g = lane >> 4;
    __bf16* lds = psh + wave * (16 * 32);

    // cooperative async-staging assignments (256 threads, 16B chunks)
    int kr = tid >> 3, kc = tid & 7;            // K: key row 0..31, chunk 0..7
    int vd = tid >> 2, vc = tid & 3;            // V: dim row 0..63, chunk 0..3
    const __bf16* kg = Kbf + ((size_t)head * N_TOK + kr) * HD + kc * 8;
    const __bf16* vg = Vtbf + ((size_t)head * HD + vd) * N_TOK + vc * 8;
    uint32_t klds[2], vlds[2];
#pragma unroll
    for (int b = 0; b < 2; ++b) {
        klds[b] = lds_off(&kbuf[b][kr * HD + kc * 8]);
        vlds[b] = lds_off(&vbuf[b][vd * 32 + vc * 8]);
    }

    // Q fragments (K = 0..31, 32..63), rows row0..row0+15, pre-scaled by 1/8
    const __bf16* qrow = Qbf + ((size_t)head * N_TOK + row0 + nl) * HD;
    v16bf qa0, qa1;
#pragma unroll
    for (int e = 0; e < 8; ++e) {
        qa0[e]     = qrow[g * 16 + e];
        qa0[8 + e] = qrow[g * 16 + 8 + e];
        qa1[e]     = qrow[32 + g * 16 + e];
        qa1[8 + e] = qrow[40 + g * 16 + e];
    }

    int si[8];
#pragma unroll
    for (int r = 0; r < 8; ++r) si[r] = steps[row0 + g * 8 + r];

    v8f o0 = {}, o1 = {}, o2 = {}, o3 = {};
    float mr[8], lr[8];
#pragma unroll
    for (int r = 0; r < 8; ++r) { mr[r] = -1e30f; lr[r] = 0.0f; }

    // prologue: stage key-block 0 into buffer 0
    async_copy_b128(klds[0], kg);
    async_copy_b128(vlds[0], vg);

    for (int i = 0; i < N_TOK / 32; ++i) {
        int c0 = i * 32;
        int b  = i & 1;
        wait_async0();            // my LDS writes for buffer b have landed
        __syncthreads();          // everyone's writes visible, prev reads done
        if (i + 1 < N_TOK / 32) { // issue next tile into the other buffer
            async_copy_b128(klds[b ^ 1], kg + (size_t)(i + 1) * 32 * HD);
            async_copy_b128(vlds[b ^ 1], vg + (i + 1) * 32);
        }

        // ---- S = Q K^T (two 16x16 tiles) from LDS ----
        const __bf16* kbase = &kbuf[b][nl * HD + g * 16];
        v16bf kb00 = load_b_contig_bf(kbase);
        v16bf kb01 = load_b_contig_bf(kbase + 32);
        v16bf kb10 = load_b_contig_bf(kbase + 16 * HD);
        v16bf kb11 = load_b_contig_bf(kbase + 16 * HD + 32);

        v8f s0 = {}, s1 = {};
        s0 = wmma_bf16(qa0, kb00, s0);
        s0 = wmma_bf16(qa1, kb01, s0);
        s1 = wmma_bf16(qa0, kb10, s1);
        s1 = wmma_bf16(qa1, kb11, s1);

        int sj0 = steps[c0 + nl];
        int sj1 = steps[c0 + 16 + nl];

        float alpha[8];
#pragma unroll
        for (int r = 0; r < 8; ++r) {
            float v0 = (sj0 < si[r]) ? -1e9f : s0[r];
            float v1 = (sj1 < si[r]) ? -1e9f : s1[r];
            float bm = fmaxf(v0, v1);
#pragma unroll
            for (int off = 8; off >= 1; off >>= 1)
                bm = fmaxf(bm, __shfl_xor(bm, off, 32));
            float mn = fmaxf(mr[r], bm);
            float al = __expf(mr[r] - mn);
            float p0 = __expf(v0 - mn);
            float p1 = __expf(v1 - mn);
            float rs = p0 + p1;
#pragma unroll
            for (int off = 8; off >= 1; off >>= 1)
                rs += __shfl_xor(rs, off, 32);
            lr[r] = lr[r] * al + rs;
            mr[r] = mn;
            alpha[r] = al;
            int m = g * 8 + r;                     // P -> LDS, 16x32 row-major
            lds[m * 32 + nl]      = (__bf16)p0;
            lds[m * 32 + 16 + nl] = (__bf16)p1;
        }
#pragma unroll
        for (int r = 0; r < 8; ++r) {
            o0[r] *= alpha[r]; o1[r] *= alpha[r];
            o2[r] *= alpha[r]; o3[r] *= alpha[r];
        }

        asm volatile("s_wait_dscnt 0" ::: "memory");   // same-wave LDS RAW

        v16bf pa;                                      // P as A fragment
        const __bf16* prow = lds + nl * 32;
#pragma unroll
        for (int e = 0; e < 8; ++e) {
            pa[e]     = prow[g * 16 + e];
            pa[8 + e] = prow[g * 16 + 8 + e];
        }

        // ---- O += P V (four 16x16 output tiles) from LDS ----
        const __bf16* vbase = &vbuf[b][nl * 32 + g * 16];
        o0 = wmma_bf16(pa, load_b_contig_bf(vbase),           o0);
        o1 = wmma_bf16(pa, load_b_contig_bf(vbase + 16 * 32), o1);
        o2 = wmma_bf16(pa, load_b_contig_bf(vbase + 32 * 32), o2);
        o3 = wmma_bf16(pa, load_b_contig_bf(vbase + 48 * 32), o3);
    }

    // normalize and emit ctx (bf16, [N][512])
#pragma unroll
    for (int r = 0; r < 8; ++r) {
        float inv = 1.0f / lr[r];
        int row = row0 + g * 8 + r;
        size_t base = (size_t)row * EDIM + head * HD + nl;
        ctxbf[base]      = (__bf16)(o0[r] * inv);
        ctxbf[base + 16] = (__bf16)(o1[r] * inv);
        ctxbf[base + 32] = (__bf16)(o2[r] * inv);
        ctxbf[base + 48] = (__bf16)(o3[r] * inv);
    }
}

// ============================================================================
// Kernel 3: status + time embeddings -> combined[:, 512:896] (bf16)
// ============================================================================
__global__ void embed_kernel(const int* __restrict__ derived_mask,
                             const int* __restrict__ steps,
                             const float* __restrict__ status_embed,
                             __bf16* __restrict__ combined)   // [N][896]
{
    int i = blockIdx.x;
    int t = threadIdx.x;                     // 128 threads
    int mk = derived_mask[i];
    int st = steps[i]; st = st < 0 ? 0 : (st > 99 ? 99 : st);
    __bf16* row = combined + (size_t)i * 896;

    row[512 + t] = (__bf16)status_embed[mk * 128 + t];
#pragma unroll
    for (int d = t; d < 256; d += 128) {
        int j = d >> 1;
        float div = __expf((float)(2 * j) * (-9.210340371976184f / 256.0f));
        float ang = (float)st * div;
        float v = (d & 1) ? __cosf(ang) : __sinf(ang);
        row[640 + d] = (__bf16)v;
    }
}

// ============================================================================
// Kernel 4: x_att = ctx @ out_proj_w^T + b -> combined[:, 0:512] (bf16)
//   4 waves/block, 4 col-tiles per wave
// ============================================================================
__global__ void outproj_gemm_kernel(const __bf16* __restrict__ ctxbf,
                                    const float* __restrict__ w,
                                    const float* __restrict__ bias,
                                    __bf16* __restrict__ combined)
{
    int lane = threadIdx.x & 31;
    int wave = threadIdx.x >> 5;
    int row0 = blockIdx.x * 16;
    int nl = lane & 15, g = lane >> 4;

    v8f acc[4] = {v8f{}, v8f{}, v8f{}, v8f{}};
#pragma unroll 2
    for (int k0 = 0; k0 < EDIM; k0 += 32) {
        v16bf a = load_a_bf(ctxbf, EDIM, row0, k0, lane);
#pragma unroll
        for (int t = 0; t < 4; ++t) {
            int col0 = (blockIdx.y * 16 + wave * 4 + t) * 16;   // 0..511
            v16bf b = load_b_w_f32(w, EDIM, col0, k0, lane);
            acc[t] = wmma_bf16(a, b, acc[t]);
        }
    }
#pragma unroll
    for (int t = 0; t < 4; ++t) {
        int c = (blockIdx.y * 16 + wave * 4 + t) * 16 + nl;
        float bv = bias[c];
#pragma unroll
        for (int r = 0; r < 8; ++r) {
            int row = row0 + g * 8 + r;
            combined[(size_t)row * 896 + c] = (__bf16)(acc[t][r] + bv);
        }
    }
}

// ============================================================================
// Kernel 5: h = combined @ mlp_w^T + b ; LayerNorm ; ReLU ; + x -> out (f32)
//   one block = 16 rows x 512 cols (8 waves x 4 tiles), K = 896
// ============================================================================
__global__ void mlp_ln_kernel(const __bf16* __restrict__ combined,
                              const float* __restrict__ w,      // [512][896]
                              const float* __restrict__ bias,
                              const float* __restrict__ gamma,
                              const float* __restrict__ beta,
                              const float* __restrict__ x,
                              float* __restrict__ out)
{
    __shared__ float hbuf[16 * 512];
    __shared__ float psum[256];
    __shared__ float psq[256];
    __shared__ float mu_s[16], rstd_s[16];

    int lane = threadIdx.x & 31;
    int wave = threadIdx.x >> 5;
    int row0 = blockIdx.x * 16;
    int nl = lane & 15, g = lane >> 4;

    v8f acc[4] = {v8f{}, v8f{}, v8f{}, v8f{}};
    for (int k0 = 0; k0 < 896; k0 += 32) {
        v16bf a = load_a_bf(combined, 896, row0, k0, lane);
        if (k0 + 32 < 896)
            __builtin_prefetch(combined + (size_t)(row0 + nl) * 896 + k0 + 32, 0, 3);
#pragma unroll
        for (int t = 0; t < 4; ++t) {
            int col0 = (wave * 4 + t) * 16;
            v16bf b = load_b_w_f32(w, 896, col0, k0, lane);
            acc[t] = wmma_bf16(a, b, acc[t]);
        }
    }
#pragma unroll
    for (int t = 0; t < 4; ++t) {
        int c = (wave * 4 + t) * 16 + nl;
        float bv = bias[c];
#pragma unroll
        for (int r = 0; r < 8; ++r)
            hbuf[(g * 8 + r) * 512 + c] = acc[t][r] + bv;
    }
    __syncthreads();

    {
        int tid = threadIdx.x;
        int row = tid >> 4, seg = tid & 15;
        const float* hp = hbuf + row * 512 + seg * 32;
        float s = 0.f, q = 0.f;
#pragma unroll
        for (int i = 0; i < 32; ++i) { float v = hp[i]; s += v; q += v * v; }
        psum[tid] = s; psq[tid] = q;
    }
    __syncthreads();
    if (threadIdx.x < 16) {
        int row = threadIdx.x;
        float s = 0.f, q = 0.f;
#pragma unroll
        for (int i = 0; i < 16; ++i) { s += psum[row * 16 + i]; q += psq[row * 16 + i]; }
        float mu = s * (1.0f / 512.0f);
        float var = q * (1.0f / 512.0f) - mu * mu;
        mu_s[row] = mu;
        rstd_s[row] = rsqrtf(var + 1e-5f);
    }
    __syncthreads();
    {
        int tid = threadIdx.x;
        int row = tid >> 4;
        float mu = mu_s[row], rstd = rstd_s[row];
        size_t gbase = (size_t)(row0 + row) * EDIM;
        for (int col = (tid & 15); col < 512; col += 16) {
            float h = hbuf[row * 512 + col];
            float v = (h - mu) * rstd * gamma[col] + beta[col];
            out[gbase + col] = x[gbase + col] + fmaxf(v, 0.0f);
        }
    }
}

// ============================================================================
extern "C" void kernel_launch(void* const* d_in, const int* in_sizes, int n_in,
                              void* d_out, int out_size, void* d_ws, size_t ws_size,
                              hipStream_t stream) {
    const float* x            = (const float*)d_in[0];
    const int*   derived_mask = (const int*)d_in[1];
    const int*   step_numbers = (const int*)d_in[2];
    const float* status_embed = (const float*)d_in[3];
    const float* in_proj_w    = (const float*)d_in[4];
    const float* in_proj_b    = (const float*)d_in[5];
    const float* out_proj_w   = (const float*)d_in[6];
    const float* out_proj_b   = (const float*)d_in[7];
    const float* mlp_w        = (const float*)d_in[8];
    const float* mlp_b        = (const float*)d_in[9];
    const float* ln_gamma     = (const float*)d_in[10];
    const float* ln_beta      = (const float*)d_in[11];
    float* out = (float*)d_out;

    char* ws = (char*)d_ws;
    const size_t szQKV = (size_t)NHEAD * N_TOK * HD * sizeof(__bf16); // 4 MiB each
    __bf16* Qbf      = (__bf16*)(ws);
    __bf16* Kbf      = (__bf16*)(ws + szQKV);
    __bf16* Vtbf     = (__bf16*)(ws + 2 * szQKV);
    __bf16* ctxbf    = (__bf16*)(ws + 3 * szQKV);
    __bf16* combined = (__bf16*)(ws + 3 * szQKV + (size_t)N_TOK * EDIM * sizeof(__bf16));

    qkv_gemm_kernel<<<dim3(N_TOK / 16, 6), dim3(128), 0, stream>>>(
        x, in_proj_w, in_proj_b, Qbf, Kbf, Vtbf);

    attention_kernel<<<dim3(256), dim3(256), 0, stream>>>(
        Qbf, Kbf, Vtbf, step_numbers, ctxbf);

    embed_kernel<<<dim3(N_TOK), dim3(128), 0, stream>>>(
        derived_mask, step_numbers, status_embed, combined);

    outproj_gemm_kernel<<<dim3(N_TOK / 16, 2), dim3(128), 0, stream>>>(
        ctxbf, out_proj_w, out_proj_b, combined);

    mlp_ln_kernel<<<dim3(N_TOK / 16), dim3(256), 0, stream>>>(
        combined, mlp_w, mlp_b, ln_gamma, ln_beta, x, out);
}
